// Network_6631429505475
// MI455X (gfx1250) — matrix operand order repaired and checked by reading, not compiled, not run on hardware
//
#include <hip/hip_runtime.h>
#include <hip/hip_bf16.h>

#define FD 128            // feature dim (fixed by problem)

typedef __attribute__((ext_vector_type(16))) __bf16 v16bf;
typedef __attribute__((ext_vector_type(8)))  float  v8f;

// ---------------------------------------------------------------------------
// Fused GEMM:  out[M,128] = A[M,128] @ W[128,128]^T + bias
// bf16 WMMA (f32 accumulate). Optionally accumulates per-column BN statistics.
// W is staged global->LDS with gfx1250 async-to-LDS copies (ASYNCcnt), then
// converted once to a bf16 LDS image used for all B fragments.
// Block = 256 threads = 8 waves; block covers 128 rows; wave covers 16 rows.
// ---------------------------------------------------------------------------
__global__ __launch_bounds__(256)
void k_gemm_xwT(const float* __restrict__ A, const float* __restrict__ W,
                const float* __restrict__ bias, float* __restrict__ out, int M,
                float* __restrict__ bnSum, float* __restrict__ bnSumSq)
{
    __shared__ float  Wf[FD * FD];                 // 64 KB f32 staging
    __shared__ __bf16 Wl[FD * FD];                 // 32 KB bf16 image
    const int tid = threadIdx.x;

    // Async copy W (64 KB) into LDS: 256 threads x 16 x 16B.
    #pragma unroll
    for (int it = 0; it < 16; ++it) {
        const int fi = (it * 256 + tid) * 4;       // float index, 16B chunk
        unsigned long long gp = (unsigned long long)(const void*)(W + fi);
        unsigned ldsOff = (unsigned)(size_t)(&Wf[fi]);   // low 32 bits = LDS offset
        asm volatile("global_load_async_to_lds_b128 %0, %1, off"
                     :: "v"(ldsOff), "v"(gp) : "memory");
    }
    asm volatile("s_wait_asynccnt 0" ::: "memory");
    __syncthreads();

    // One-shot f32 -> bf16 conversion.
    for (int i = tid * 4; i < FD * FD; i += 256 * 4) {
        Wl[i + 0] = (__bf16)Wf[i + 0];  Wl[i + 1] = (__bf16)Wf[i + 1];
        Wl[i + 2] = (__bf16)Wf[i + 2];  Wl[i + 3] = (__bf16)Wf[i + 3];
    }
    __syncthreads();

    const int wave = tid >> 5;
    const int lane = tid & 31;
    const int rowBase = blockIdx.x * 128 + wave * 16;
    if (rowBase >= M) return;                      // M is a multiple of 16

    const int half = lane >> 4;                    // 0: lanes 0-15, 1: lanes 16-31
    const int l16  = lane & 15;
    const int row  = rowBase + l16;                // A row this lane feeds

    v8f c[8];
    v8f zero = {};
    #pragma unroll
    for (int t = 0; t < 8; ++t) c[t] = zero;

    const float* __restrict__ arow = A + (size_t)row * FD;

    #pragma unroll
    for (int ks = 0; ks < 4; ++ks) {               // K = 128 = 4 x 32
        const int K0 = ks * 32;
        // A fragment, contiguous K order (matches B):
        //   lane half h, elem e -> K = K0 + 16h + e
        v16bf a;
        const float* ap = arow + K0 + half * 16;
        #pragma unroll
        for (int e = 0; e < 8; ++e) a[e]     = (__bf16)ap[e];
        #pragma unroll
        for (int e = 0; e < 8; ++e) a[8 + e] = (__bf16)ap[8 + e];

        #pragma unroll
        for (int t = 0; t < 8; ++t) {              // 8 N-tiles of 16
            const int n = t * 16 + l16;
            // B fragment: lane=column n, K = K0 + 16h .. +15 -> contiguous W row
            v16bf b = *(const v16bf*)(&Wl[n * FD + K0 + half * 16]);
            c[t] = __builtin_amdgcn_wmma_f32_16x16x32_bf16(
                       false, a, false, b, (short)0, c[t], false, false);
        }
    }

    // Epilogue: bias add, store, fused BN statistics.
    #pragma unroll
    for (int t = 0; t < 8; ++t) {
        const int n  = t * 16 + l16;
        const float bs = bias ? bias[n] : 0.0f;
        float s = 0.f, s2 = 0.f;
        #pragma unroll
        for (int i = 0; i < 8; ++i) {              // C VGPR i -> M = rowBase + 8h + i
            const float v = c[t][i] + bs;
            out[(size_t)(rowBase + half * 8 + i) * FD + n] = v;
            s += v;  s2 += v * v;
        }
        if (bnSum) {
            atomicAdd(bnSum   + n, s);
            atomicAdd(bnSumSq + n, s2);
        }
    }
}

// ---------------------------------------------------------------------------
// CSR construction over dst (graph reused by both layers).
// ---------------------------------------------------------------------------
__global__ void k_ones(int* __restrict__ cnt, int N)
{
    int i = blockIdx.x * blockDim.x + threadIdx.x;
    if (i < N) cnt[i] = 1;                         // the self-loop edge
}
__global__ void k_count(const int* __restrict__ dst, int* __restrict__ cnt, int E)
{
    int i = blockIdx.x * blockDim.x + threadIdx.x;
    if (i < E) atomicAdd(cnt + dst[i], 1);
}
// Single-workgroup exclusive scan (N ~ 50k): chunk-per-thread + LDS scan.
__global__ __launch_bounds__(1024)
void k_scan(const int* __restrict__ cnt, int* __restrict__ rowStart, int N)
{
    __shared__ int tsum[1024];
    const int t = threadIdx.x;
    const int chunk = (N + 1023) >> 10;
    const int beg = t * chunk;
    const int end = min(beg + chunk, N);
    int s = 0;
    for (int i = beg; i < end; ++i) s += cnt[i];
    tsum[t] = s;
    __syncthreads();
    for (int off = 1; off < 1024; off <<= 1) {     // Hillis-Steele inclusive scan
        int v = (t >= off) ? tsum[t - off] : 0;
        __syncthreads();
        tsum[t] += v;
        __syncthreads();
    }
    int run = (t == 0) ? 0 : tsum[t - 1];
    for (int i = beg; i < end; ++i) { rowStart[i] = run; run += cnt[i]; }
    if (t == 1023) rowStart[N] = run;              // total = E + N
}
__global__ void k_fill(const int* __restrict__ dst, const int* __restrict__ rowStart,
                       int* __restrict__ cursor, int* __restrict__ edgeIdx, int E, int N)
{
    long long i = (long long)blockIdx.x * blockDim.x + threadIdx.x;
    if (i >= (long long)E + N) return;
    const int d = (i < E) ? dst[i] : (int)(i - E);
    const int pos = rowStart[d] + atomicAdd(cursor + d, 1);
    edgeIdx[pos] = (int)i;                         // i >= E encodes the self loop
}

// ---------------------------------------------------------------------------
// Gather-style aggregation: one wave per destination node; lane owns 4 cols.
// Zero float atomics; inv_deg normalization folded in (deg = row extent).
// Prefetches the next edge's source row (global_prefetch) to hide L2 latency.
// ---------------------------------------------------------------------------
__global__ __launch_bounds__(256)
void k_aggregate(const float* __restrict__ srcTab, const int* __restrict__ node_id,
                 const int* __restrict__ src_in, const int* __restrict__ etype,
                 const float* __restrict__ rel, const int* __restrict__ rowStart,
                 const int* __restrict__ edgeIdx, float* __restrict__ agg,
                 int E, int N, int selfRel)
{
    const int wave = threadIdx.x >> 5;
    const int lane = threadIdx.x & 31;
    const int d = blockIdx.x * 8 + wave;
    if (d >= N) return;

    const int beg = rowStart[d];
    const int end = rowStart[d + 1];
    float4 acc = make_float4(0.f, 0.f, 0.f, 0.f);

    // Software-pipelined: resolve edge j+1's source row and prefetch it while
    // consuming edge j.
    int idx  = edgeIdx[beg];
    int s    = (idx < E) ? src_in[idx] : d;
    int t    = (idx < E) ? etype[idx]  : selfRel;
    int srow = node_id ? node_id[s] : s;

    for (int j = beg; j < end; ++j) {
        const int curRow = srow, curT = t;
        if (j + 1 < end) {
            idx  = edgeIdx[j + 1];
            s    = (idx < E) ? src_in[idx] : d;
            t    = (idx < E) ? etype[idx]  : selfRel;
            srow = node_id ? node_id[s] : s;
            __builtin_prefetch(srcTab + (size_t)srow * FD + lane * 4, 0, 0);
        }
        const float4 v = *((const float4*)(srcTab + (size_t)curRow * FD) + lane);
        const float4 r = *((const float4*)(rel    + (size_t)curT   * FD) + lane);
        acc.x += v.x * r.x;  acc.y += v.y * r.y;
        acc.z += v.z * r.z;  acc.w += v.w * r.w;
    }

    const float inv = 1.0f / (float)(end - beg);   // deg >= 1 (self loop)
    acc.x *= inv; acc.y *= inv; acc.z *= inv; acc.w *= inv;
    *((float4*)(agg + (size_t)d * FD) + lane) = acc;
}

// ---------------------------------------------------------------------------
// Tiny relation GEMMs (kept f32 VALU — 0.1% of total work, full precision).
// ---------------------------------------------------------------------------
__global__ void k_rel_base(const float* __restrict__ rel_wt, const float* __restrict__ emb_e,
                           float* __restrict__ out, int R, int KB)
{
    int idx = blockIdx.x * blockDim.x + threadIdx.x;
    if (idx >= R * FD) return;
    int r = idx >> 7, n = idx & 127;
    float acc = 0.f;
    for (int k = 0; k < KB; ++k) acc += rel_wt[r * KB + k] * emb_e[k * FD + n];
    out[idx] = acc;
}
__global__ void k_rel_next(const float* __restrict__ rel, const float* __restrict__ w,
                           float* __restrict__ out, int R)
{
    int idx = blockIdx.x * blockDim.x + threadIdx.x;
    if (idx >= R * FD) return;
    int r = idx >> 7, n = idx & 127;
    float acc = 0.f;
    for (int k = 0; k < FD; ++k) acc += rel[r * FD + k] * w[k * FD + n];
    out[idx] = acc;
}

// ---------------------------------------------------------------------------
// BatchNorm: fold sums into per-column scale/shift, then apply (opt. ReLU).
// ---------------------------------------------------------------------------
__global__ void k_bn_params(const float* __restrict__ sum, const float* __restrict__ sumsq,
                            const float* __restrict__ g, const float* __restrict__ b,
                            float* __restrict__ scale, float* __restrict__ shift, float invM)
{
    int n = threadIdx.x;                           // 128 threads
    float mu  = sum[n] * invM;
    float var = sumsq[n] * invM - mu * mu;         // biased variance (matches jnp.var)
    float sc  = g[n] * rsqrtf(var + 1e-5f);
    scale[n] = sc;
    shift[n] = b[n] - mu * sc;
}
__global__ void k_bn_apply(const float* __restrict__ x, const float* __restrict__ scale,
                           const float* __restrict__ shift, float* __restrict__ out,
                           int count, int relu)
{
    int i = blockIdx.x * blockDim.x + threadIdx.x;
    if (i >= count) return;
    int n = i & 127;
    float v = x[i] * scale[n] + shift[n];
    if (relu) v = fmaxf(v, 0.f);
    out[i] = v;
}

// ---------------------------------------------------------------------------
// Host-side orchestration (graph-capture safe: kernels + hipMemsetAsync only).
// ---------------------------------------------------------------------------
extern "C" void kernel_launch(void* const* d_in, const int* in_sizes, int n_in,
                              void* d_out, int out_size, void* d_ws, size_t ws_size,
                              hipStream_t stream)
{
    const int*   node_id = (const int*)  d_in[0];
    const int*   src_in  = (const int*)  d_in[1];
    const int*   dst_in  = (const int*)  d_in[2];
    const int*   etype   = (const int*)  d_in[3];
    const float* emb_h   = (const float*)d_in[4];
    const float* emb_e   = (const float*)d_in[5];
    const float* W_lin   = (const float*)d_in[6];
    const float* b_lin   = (const float*)d_in[7];
    const float* rel_wt  = (const float*)d_in[8];
    const float* w_rel   = (const float*)d_in[9];
    const float* W_cells = (const float*)d_in[10];
    const float* gamma   = (const float*)d_in[11];
    const float* beta    = (const float*)d_in[12];

    const int N       = in_sizes[0];
    const int E       = in_sizes[1];
    const int NUM_ENT = in_sizes[4] / FD;
    const int KB      = in_sizes[5] / FD;          // NUM_BASE_R (100)
    const int NUM_REL = in_sizes[8] / KB;          // 401

    // Workspace carve-up.
    float* ws = (float*)d_ws;
    size_t o = 0;
    float* all_ent  = ws + o;  o += (size_t)NUM_ENT * FD;
    float* agg      = ws + o;  o += (size_t)N * FD;
    float* ent_lin  = ws + o;  o += (size_t)N * FD;
    float* ent0     = ws + o;  o += (size_t)N * FD;
    float* rel0     = ws + o;  o += (size_t)NUM_REL * FD;
    float* rel1     = ws + o;  o += (size_t)NUM_REL * FD;
    float* bn_sum   = ws + o;  o += 128;           // bn_sum & bn_sumsq contiguous
    float* bn_sumsq = ws + o;  o += 128;
    float* bn_scale = ws + o;  o += 128;
    float* bn_shift = ws + o;  o += 128;
    int*   cnt      = (int*)(ws + o);              // doubles as fill cursor
    int*   rowStart = cnt + N;                     // N+1 entries
    int*   edgeIdx  = rowStart + (N + 1);          // E+N entries

    float* out_ent = (float*)d_out;
    float* out_rel = (float*)d_out + (size_t)N * FD;

    const int relElems  = NUM_REL * FD;
    const int relBlocks = (relElems + 255) / 256;
    const long long totEdges = (long long)E + N;
    const int entElems  = N * FD;
    const int applyBlk  = (entElems + 255) / 256;
    const int aggBlocks = (N + 7) / 8;

    // Relation chain: rel0 = rel_wt@emb_e; rel1 = rel0@w_rel; output rel = rel1@w_rel.
    k_rel_base<<<relBlocks, 256, 0, stream>>>(rel_wt, emb_e, rel0, NUM_REL, KB);
    k_rel_next<<<relBlocks, 256, 0, stream>>>(rel0, w_rel, rel1, NUM_REL);
    k_rel_next<<<relBlocks, 256, 0, stream>>>(rel1, w_rel, out_rel, NUM_REL);

    // all_ent = emb_h @ W_lin^T + b_lin   (WMMA)
    k_gemm_xwT<<<(NUM_ENT + 127) / 128, 256, 0, stream>>>(
        emb_h, W_lin, b_lin, all_ent, NUM_ENT, nullptr, nullptr);

    // Build CSR over destinations once (shared by both layers).
    k_ones <<<(N + 255) / 256, 256, 0, stream>>>(cnt, N);
    k_count<<<(E + 255) / 256, 256, 0, stream>>>(dst_in, cnt, E);
    k_scan <<<1, 1024, 0, stream>>>(cnt, rowStart, N);
    hipMemsetAsync(cnt, 0, (size_t)N * sizeof(int), stream);     // reuse as cursor
    k_fill <<<(int)((totEdges + 255) / 256), 256, 0, stream>>>(
        dst_in, rowStart, cnt, edgeIdx, E, N);

    // ---- Layer 0 ----
    k_aggregate<<<aggBlocks, 256, 0, stream>>>(all_ent, node_id, src_in, etype,
                                               rel0, rowStart, edgeIdx, agg,
                                               E, N, NUM_REL - 1);
    hipMemsetAsync(bn_sum, 0, 2 * 128 * sizeof(float), stream);
    k_gemm_xwT<<<(N + 127) / 128, 256, 0, stream>>>(
        agg, W_cells, nullptr, ent_lin, N, bn_sum, bn_sumsq);
    k_bn_params<<<1, 128, 0, stream>>>(bn_sum, bn_sumsq, gamma, beta,
                                       bn_scale, bn_shift, 1.0f / (float)N);
    k_bn_apply<<<applyBlk, 256, 0, stream>>>(ent_lin, bn_scale, bn_shift,
                                             ent0, entElems, 0);   // no ReLU on layer 0

    // ---- Layer 1 ----
    k_aggregate<<<aggBlocks, 256, 0, stream>>>(ent0, nullptr, src_in, etype,
                                               rel1, rowStart, edgeIdx, agg,
                                               E, N, NUM_REL - 1);
    hipMemsetAsync(bn_sum, 0, 2 * 128 * sizeof(float), stream);
    k_gemm_xwT<<<(N + 127) / 128, 256, 0, stream>>>(
        agg, W_cells + FD * FD, nullptr, ent_lin, N, bn_sum, bn_sumsq);
    k_bn_params<<<1, 128, 0, stream>>>(bn_sum, bn_sumsq, gamma, beta,
                                       bn_scale, bn_shift, 1.0f / (float)N);
    k_bn_apply<<<applyBlk, 256, 0, stream>>>(ent_lin, bn_scale, bn_shift,
                                             out_ent, entElems, 1); // ReLU on layer 1
}
